// AttentionLayer_73452530696414
// MI455X (gfx1250) — matrix operand order repaired and checked by reading, not compiled, not run on hardware
//
#include <hip/hip_runtime.h>
#include <hip/hip_bf16.h>

#define T_SNAP 128
#define NNODES 2048
#define DDIM   256
#define WWIN   3
#define WT_STRIDE 264                      // 256 + 8 halves pad: conflict-free LDS, keeps 16B align
#define WT_HALVES (DDIM * WT_STRIDE)       // 67584 halves = 135168 B
#define WT_BYTES  (WT_HALVES * 2)
#define SMEM_BYTES WT_BYTES

typedef _Float16 half8  __attribute__((ext_vector_type(8)));
typedef _Float16 half16 __attribute__((ext_vector_type(16)));
typedef float    f32x8  __attribute__((ext_vector_type(8)));
typedef int      vint4  __attribute__((vector_size(16)));   // async builtin's expected pointee

#define AS_GLOBAL __attribute__((address_space(1)))
#define AS_LDS    __attribute__((address_space(3)))

// Scheduler control: force DS-load / WMMA interleave (0x100 = DS read, 0x8 = MFMA/WMMA).
#if __has_builtin(__builtin_amdgcn_sched_group_barrier)
#define SGB(mask, n) __builtin_amdgcn_sched_group_barrier(mask, n, 0)
#else
#define SGB(mask, n)
#endif

// Hardware V_TANH_F32 (CDNA5 TRANS op); branchless fallback if builtin absent.
__device__ __forceinline__ float fast_tanhf(float x)
{
#if __has_builtin(__builtin_amdgcn_tanhf)
    return __builtin_amdgcn_tanhf(x);
#elif __has_builtin(__builtin_amdgcn_tanh_f32)
    return __builtin_amdgcn_tanh_f32(x);
#else
    x = fminf(fmaxf(x, -15.f), 15.f);
    const float e2 = __expf(2.0f * x);
    return (e2 - 1.0f) / (e2 + 1.0f);
#endif
}

// B-fragment (32x16 f16): lane holds K 0..7 and 16..23 at column n=lane&15
// (+8 K for high half-wave). Two ds_load_b128 from padded LDS image.
__device__ __forceinline__ half16 loadB(const _Float16* w, int kc)
{
    const half8 lo = *(const half8*)(w + kc * 32);
    const half8 hi = *(const half8*)(w + kc * 32 + 16);
    return __builtin_shufflevector(lo, hi,
        0,1,2,3,4,5,6,7,8,9,10,11,12,13,14,15);
}

// ---------------------------------------------------------------------------
// Stage 0: one-time W (f32 [k][e]) -> WtG (f16, transposed [e][k], padded rows).
// ---------------------------------------------------------------------------
__global__ __launch_bounds__(256)
void prep_kernel(const float* __restrict__ W, _Float16* __restrict__ WtG)
{
    const int idx = blockIdx.x * 256 + threadIdx.x;   // 65536 threads, coalesced read
    const int k = idx >> 8;
    const int e = idx & 255;
    WtG[e * WT_STRIDE + k] = (_Float16)W[idx];
}

// ---------------------------------------------------------------------------
// Stage A: s[t,n] = sum_e proj[e] * tanh( sum_d x[t,n,d] * W[d,e] )
// Grid: (T, N/128). 8 waves/WG; wave w owns node tile nBase + 16*w.
// 16 e-tiles x 8 k-chunks = 128 x v_wmma_f32_16x16x32_f16 per wave, as two
// independent accumulation chains with sched_group_barrier-pipelined B loads.
// ---------------------------------------------------------------------------
__global__ __launch_bounds__(256)
void score_kernel(const float* __restrict__ X, const _Float16* __restrict__ WtG,
                  const float* __restrict__ proj, float* __restrict__ S)
{
    extern __shared__ char smemRaw[];
    _Float16* Wt = (_Float16*)smemRaw;                 // [e*WT_STRIDE + k]

    const int t     = blockIdx.x;
    const int nBase = blockIdx.y * 128;
    const int tid   = threadIdx.x;

    // Stage WtG (global f16) -> LDS via the CDNA5 async-copy path when available.
#if __has_builtin(__builtin_amdgcn_global_load_async_to_lds_b128)
    {
        const char* gsrc = (const char*)WtG;
        char*       ldst = (char*)smemRaw;
#pragma unroll 1
        for (int it = 0; it < WT_BYTES / (256 * 16); ++it) {   // 33 x b128 per thread
            const int ofs = (tid + it * 256) * 16;
            __builtin_amdgcn_global_load_async_to_lds_b128(
                (AS_GLOBAL vint4*)(gsrc + ofs), (AS_LDS vint4*)(ldst + ofs), 0, 0);
        }
#if __has_builtin(__builtin_amdgcn_s_wait_asynccnt)
        __builtin_amdgcn_s_wait_asynccnt(0);
#else
        asm volatile("s_wait_asynccnt 0" ::: "memory");
#endif
    }
#else
    {
        half8*       dstv = (half8*)Wt;
        const half8* srcv = (const half8*)WtG;
        for (int i = tid; i < WT_HALVES / 8; i += 256) dstv[i] = srcv[i];
    }
#endif
    __syncthreads();

    const int lane  = tid & 31;
    const int wv    = tid >> 5;
    const int m     = lane & 15;                 // A row within tile
    const int hi    = lane >> 4;                 // lane group
    const int kHiA  = hi << 3;                   // A frag: +8 for high lanes
    const int kHiB  = hi << 3;                   // B frag: +8 for high lanes
    const int node  = nBase + wv * 16 + m;

    const float* xrow = X + ((size_t)t * NNODES + node) * DDIM;

    // A fragments for all 8 K-chunks (K=32 each), f32 -> f16 on load.
    half16 afrag[8];
#pragma unroll
    for (int kc = 0; kc < 8; ++kc) {
        const float4* q = (const float4*)(xrow + kc * 32 + kHiA);
        const float4 v0 = q[0], v1 = q[1];       // K = base .. base+7
        const float4 v2 = q[4], v3 = q[5];       // K = base+16 .. base+23
        half16 a;
        a[0]=(_Float16)v0.x; a[1]=(_Float16)v0.y; a[2] =(_Float16)v0.z; a[3] =(_Float16)v0.w;
        a[4]=(_Float16)v1.x; a[5]=(_Float16)v1.y; a[6] =(_Float16)v1.z; a[7] =(_Float16)v1.w;
        a[8]=(_Float16)v2.x; a[9]=(_Float16)v2.y; a[10]=(_Float16)v2.z; a[11]=(_Float16)v2.w;
        a[12]=(_Float16)v3.x;a[13]=(_Float16)v3.y;a[14] =(_Float16)v3.z;a[15] =(_Float16)v3.w;
        afrag[kc] = a;
    }

    float sAcc[8] = {0.f,0.f,0.f,0.f,0.f,0.f,0.f,0.f};
    const int eLane = lane & 15;

#pragma unroll 1
    for (int et = 0; et < 16; et += 2) {
        const int e0 = et * 16 + eLane;
        const int e1 = e0 + 16;
        const _Float16* w0 = Wt + e0 * WT_STRIDE + kHiB;
        const _Float16* w1 = Wt + e1 * WT_STRIDE + kHiB;
        f32x8 acc0 = {0.f,0.f,0.f,0.f,0.f,0.f,0.f,0.f};
        f32x8 acc1 = {0.f,0.f,0.f,0.f,0.f,0.f,0.f,0.f};

        // Software pipeline, enforced with sched_group_barrier:
        //   DSx4 (kc=0) -> { DSx4 (kc+1), WMMAx2 (kc) } x7 -> WMMAx2 (kc=7)
        half16 b0 = loadB(w0, 0);
        half16 b1 = loadB(w1, 0);
        SGB(0x100, 4);                               // kc=0 B loads
#pragma unroll
        for (int kc = 0; kc < 8; ++kc) {
            half16 n0 = b0, n1 = b1;
            if (kc < 7) {
                n0 = loadB(w0, kc + 1);
                n1 = loadB(w1, kc + 1);
                SGB(0x100, 4);                       // kc+1 loads BEFORE kc's WMMAs
            }
            acc0 = __builtin_amdgcn_wmma_f32_16x16x32_f16(
                false, afrag[kc], false, b0, (short)0, acc0, false, false);
            acc1 = __builtin_amdgcn_wmma_f32_16x16x32_f16(
                false, afrag[kc], false, b1, (short)0, acc1, false, false);
            SGB(0x8, 2);                             // the two WMMAs for kc
            b0 = n0; b1 = n1;
        }
        // proj from global (L2-resident): VMEM, keeps DS groups clean.
        const float p0 = proj[e0];
        const float p1 = proj[e1];
#pragma unroll
        for (int r = 0; r < 8; ++r)
            sAcc[r] += p0 * fast_tanhf(acc0[r]) + p1 * fast_tanhf(acc1[r]);
    }

    // Reduce over the 16 e-lanes within each half-wave.
#pragma unroll
    for (int r = 0; r < 8; ++r) {
        float v = sAcc[r];
#pragma unroll
        for (int msk = 1; msk < 16; msk <<= 1) v += __shfl_xor(v, msk, 32);
        sAcc[r] = v;
    }
    if (eLane == 0) {
        const int mrow = hi * 8;                 // lane0 -> rows 0-7, lane16 -> rows 8-15
#pragma unroll
        for (int r = 0; r < 8; ++r)
            S[(size_t)t * NNODES + nBase + wv * 16 + mrow + r] = sAcc[r];
    }
}

// ---------------------------------------------------------------------------
// Stage B: in-place softmax over the node axis, one block per snapshot t.
// ---------------------------------------------------------------------------
__global__ __launch_bounds__(256)
void softmax_kernel(float* __restrict__ S)
{
    __shared__ float red[8];
    const int t = blockIdx.x;
    float* row = S + (size_t)t * NNODES;
    const int tid = threadIdx.x, lane = tid & 31, wv = tid >> 5;

    float mx = -3.0e38f;
    for (int n = tid; n < NNODES; n += 256) mx = fmaxf(mx, row[n]);
#pragma unroll
    for (int m = 16; m >= 1; m >>= 1) mx = fmaxf(mx, __shfl_xor(mx, m, 32));
    if (lane == 0) red[wv] = mx;
    __syncthreads();
    mx = red[lane & 7];
#pragma unroll
    for (int m = 4; m >= 1; m >>= 1) mx = fmaxf(mx, __shfl_xor(mx, m, 32));

    float lsum = 0.f;
    for (int n = tid; n < NNODES; n += 256) {
        const float e = __expf(row[n] - mx);
        row[n] = e;
        lsum += e;
    }
#pragma unroll
    for (int m = 16; m >= 1; m >>= 1) lsum += __shfl_xor(lsum, m, 32);
    __syncthreads();
    if (lane == 0) red[wv] = lsum;
    __syncthreads();
    lsum = red[lane & 7];
#pragma unroll
    for (int m = 4; m >= 1; m >>= 1) lsum += __shfl_xor(lsum, m, 32);

    const float inv = 1.0f / lsum;
    for (int n = tid; n < NNODES; n += 256) row[n] *= inv;
}

// ---------------------------------------------------------------------------
// Stage C: out[i] = x[i] (i<w) ; out[i] = sum_{j<w} p[i-w+j]*x[i-w+j] (i>=w).
// One thread per float4 column; sliding 3-deep ring -> X read exactly once.
// ---------------------------------------------------------------------------
__global__ __launch_bounds__(256)
void out_kernel(const float* __restrict__ X, const float* __restrict__ P,
                float* __restrict__ out)
{
    const int col = blockIdx.x * 256 + threadIdx.x;     // 0 .. N*D/4-1
    const int n   = col >> 6;
    const int d4  = (col & 63) << 2;
    const size_t colOff = (size_t)n * DDIM + d4;
    const float4* xq = (const float4*)(X + colOff);
    float4*       oq = (float4*)(out + colOff);
    const size_t stepQ = (size_t)NNODES * DDIM / 4;

    float4 xw[WWIN];
    float  pw[WWIN];
#pragma unroll
    for (int t = 0; t < WWIN; ++t) {
        const float4 v = xq[(size_t)t * stepQ];
        oq[(size_t)t * stepQ] = v;                      // passthrough rows
        xw[t] = v;
        pw[t] = P[(size_t)t * NNODES + n];
    }

#pragma unroll 1
    for (int i = WWIN; i < T_SNAP; ++i) {
        float4 xn = {0.f, 0.f, 0.f, 0.f};
        float  pn = 0.f;
        if (i < T_SNAP - 1) {                           // fetch next window member early
            xn = xq[(size_t)i * stepQ];
            pn = P[(size_t)i * NNODES + n];
            if (i + 1 < T_SNAP - 1)
                __builtin_prefetch((const void*)(xq + (size_t)(i + 1) * stepQ), 0, 3);
        }
        float4 o;
        o.x = pw[0]*xw[0].x + pw[1]*xw[1].x + pw[2]*xw[2].x;
        o.y = pw[0]*xw[0].y + pw[1]*xw[1].y + pw[2]*xw[2].y;
        o.z = pw[0]*xw[0].z + pw[1]*xw[1].z + pw[2]*xw[2].z;
        o.w = pw[0]*xw[0].w + pw[1]*xw[1].w + pw[2]*xw[2].w;
        oq[(size_t)i * stepQ] = o;
        xw[0] = xw[1]; xw[1] = xw[2];
        pw[0] = pw[1]; pw[1] = pw[2];
        if (i < T_SNAP - 1) { xw[2] = xn; pw[2] = pn; }
    }
}

// ---------------------------------------------------------------------------
extern "C" void kernel_launch(void* const* d_in, const int* in_sizes, int n_in,
                              void* d_out, int out_size, void* d_ws, size_t ws_size,
                              hipStream_t stream)
{
    (void)in_sizes; (void)n_in; (void)out_size; (void)ws_size;
    const float* X    = (const float*)d_in[0];   // (T, N, D) f32
    const float* W    = (const float*)d_in[1];   // (D, D)    f32
    const float* proj = (const float*)d_in[2];   // (D, 1)    f32
    float* out = (float*)d_out;                  // (T, N, D) f32

    float*     S   = (float*)d_ws;                                   // T*N f32 (1 MB)
    _Float16*  WtG = (_Float16*)((char*)d_ws + (size_t)T_SNAP * NNODES * 4);

    prep_kernel<<<DDIM * DDIM / 256, 256, 0, stream>>>(W, WtG);
    score_kernel<<<dim3(T_SNAP, NNODES / 128), 256, SMEM_BYTES, stream>>>(X, WtG, proj, S);
    softmax_kernel<<<T_SNAP, 256, 0, stream>>>(S);
    out_kernel<<<(NNODES * DDIM / 4) / 256, 256, 0, stream>>>(X, S, out);
}
